// RNNETS_64321430225704
// MI455X (gfx1250) — matrix operand order repaired and checked by reading, not compiled, not run on hardware
//
#include <hip/hip_runtime.h>
#include <hip/hip_bf16.h>

#define N_NODES   100000
#define N_EDGES   3200000
#define N_GRAPHS  512
#define F_IN      128
#define HIDDEN    128
#define N_CLASSES 10
#define N_TILES   (N_NODES / 16)   // 6250

typedef __attribute__((ext_vector_type(16))) _Float16 v16h;
typedef __attribute__((ext_vector_type(8)))  float    v8f;

// ---------------------------------------------------------------------------
// Workspace layout (bytes, all 256-aligned):
//   [0,           25,600,000)  hin  : f16 [N_NODES][128] = f16(x), then += edges
//   [25,600,000,  25,862,144)  g    : f32 [N_GRAPHS][128] pooled sums
//   [25,862,144,  25,927,680)  w1p  : f16 packed WMMA-B frags of conv_w1
//   [25,927,680,  25,993,216)  w2p  : f16 packed WMMA-B frags of conv_w2
// Everything L2-resident (192 MB L2); the 205M packed atomics never hit HBM.
// ---------------------------------------------------------------------------
#define OFF_G    25600000
#define OFF_W1P  25862144
#define OFF_W2P  25927680

// ---------------- hin = f16(x)  (self term of x + agg); zero pooled g ------
__global__ void init_hin_g(const float* __restrict__ x,
                           _Float16* __restrict__ hin,
                           float* __restrict__ g, int nh, int ng) {
  int i = blockIdx.x * blockDim.x + threadIdx.x;
  if (i < nh) hin[i] = (_Float16)x[i];
  if (i < ng) g[i] = 0.0f;
}

// ---------------- pack weights into WMMA B-matrix (32x16 f16) fragments ----
// B layout (K x N = 32x16): lane L holds column N = L&15; halves j=0..15 are
// K = (j>=8?16:0) + (L>>4)*8 + (j&7).  Fragment storage: frag[(kt*8+nt)] -> 32 lanes x 16 halves.
__global__ void pack_weights(const float* __restrict__ w1,
                             const float* __restrict__ w2,
                             _Float16* __restrict__ w1p,
                             _Float16* __restrict__ w2p) {
  int f    = blockIdx.x;       // 0..63 : [layer(1) | kt(2) | nt(3)]
  int lane = threadIdx.x;      // 0..31
  int layer = f >> 5;
  int fi    = f & 31;
  int kt = fi >> 3, nt = fi & 7;
  const float* W = layer ? w2 : w1;
  _Float16*    P = layer ? w2p : w1p;
  int N = lane & 15, half = lane >> 4;
  _Float16* dst = P + ((long long)fi * 32 + lane) * 16;
#pragma unroll
  for (int j = 0; j < 16; ++j) {
    int K = kt * 32 + ((j >= 8) ? 16 : 0) + half * 8 + (j & 7);
    dst[j] = (_Float16)W[K * HIDDEN + nt * 16 + N];
  }
}

// ---------------- edge scatter-add: hin[dst] += f16(x[src]) ----------------
// One wave per edge; lane q handles features [4q, 4q+4): one float4 gather,
// two GLOBAL_ATOMIC_PK_ADD_F16 (2xf16 per atomic -> 64 atomics/edge instead
// of 128 f32 atomics). Destination buffer (25.6 MB) is L2-resident.
__global__ void edge_scatter_pk(const float4* __restrict__ x4,
                                const long long* __restrict__ ei,
                                _Float16* __restrict__ hin) {
  long long idx = (long long)blockIdx.x * blockDim.x + threadIdx.x;
  if (idx >= (long long)N_EDGES * 32) return;
  int e = (int)(idx >> 5);
  int q = (int)(idx & 31);
  int src = (int)ei[e];
  int dst = (int)ei[(long long)N_EDGES + e];
  float4 v = x4[(long long)src * 32 + q];
  union { _Float16 h[2]; unsigned int u; } p0, p1;
  p0.h[0] = (_Float16)v.x; p0.h[1] = (_Float16)v.y;
  p1.h[0] = (_Float16)v.z; p1.h[1] = (_Float16)v.w;
  _Float16* base = hin + (long long)dst * 128 + q * 4;
  asm volatile("global_atomic_pk_add_f16 %0, %1, off"
               :: "v"((unsigned long long)base), "v"(p0.u) : "memory");
  asm volatile("global_atomic_pk_add_f16 %0, %1, off"
               :: "v"((unsigned long long)(base + 2)), "v"(p1.u) : "memory");
}

// ---------------- fused 2-layer node MLP (WMMA) + pooled atomic add --------
// One wave per 16-node tile. A frag (16x32 f16) per ISA layout: lane L -> row
// M=L&15, half=L>>4; VGPR v holds K = kt*32 + (v>=4?16:0) + 8*half + 2*(v&3).
// C/D layout: lane L holds column N=L&15, VGPR r holds row M = r + 8*(L>>4).
union AFrag { v16h v; unsigned int u[8]; };

__global__ void __launch_bounds__(256)
node_mlp(const _Float16* __restrict__ hin,
         const _Float16* __restrict__ w1p,
         const _Float16* __restrict__ w2p,
         const float* __restrict__ b1,
         const float* __restrict__ b2,
         const long long* __restrict__ batch,
         float* __restrict__ g) {
  __shared__ _Float16 lds[8][16][HIDDEN];   // per-wave 16x128 f16 staging (32 KB)
  const int lane = threadIdx.x & 31;
  const int wv   = threadIdx.x >> 5;
  const int wg   = blockIdx.x * 8 + wv;
  const bool active = (wg < N_TILES);
  const int tile = active ? wg : (N_TILES - 1);   // clamp; redundant waves don't store
  const long long node0 = (long long)tile * 16;
  const int M = lane & 15, half = lane >> 4;

  // --- layer-1 A fragments straight from global hin (2-half dword loads) ---
  AFrag a[4];
  const unsigned int* hrow = (const unsigned int*)(hin + (node0 + M) * HIDDEN);
#pragma unroll
  for (int kt = 0; kt < 4; ++kt)
#pragma unroll
    for (int v = 0; v < 8; ++v) {
      int k = kt * 32 + ((v >= 4) ? 16 : 0) + half * 8 + 2 * (v & 3);
      a[kt].u[v] = hrow[k >> 1];
    }

  // --- layer 1: h1 = relu(A @ W1 + b1) -> LDS (f16) ---
#pragma unroll
  for (int nt = 0; nt < 8; ++nt) {
    v8f c = {0.f, 0.f, 0.f, 0.f, 0.f, 0.f, 0.f, 0.f};
#pragma unroll
    for (int kt = 0; kt < 4; ++kt) {
      v16h b = *(const v16h*)(w1p + ((long long)(kt * 8 + nt) * 32 + lane) * 16);
      c = __builtin_amdgcn_wmma_f32_16x16x32_f16(false, a[kt].v, false, b,
                                                 (short)0, c, false, false);
    }
    float bias = b1[nt * 16 + M];
#pragma unroll
    for (int r = 0; r < 8; ++r) {
      int Mr = half ? (r + 8) : r;
      float val = c[r] + bias;
      lds[wv][Mr][nt * 16 + M] = (_Float16)(val > 0.f ? val : 0.f);
    }
  }
  __syncthreads();   // C-layout (row-per-VGPR) -> A-layout (row-per-lane) reshuffle

  // --- layer-2 A fragments from LDS ---
  const unsigned int* lrow = (const unsigned int*)(&lds[wv][M][0]);
#pragma unroll
  for (int kt = 0; kt < 4; ++kt)
#pragma unroll
    for (int v = 0; v < 8; ++v) {
      int k = kt * 32 + ((v >= 4) ? 16 : 0) + half * 8 + 2 * (v & 3);
      a[kt].u[v] = lrow[k >> 1];
    }

  // --- layer 2: h = relu(h1 @ W2 + b2); pool: g[batch[node]] += h ---
#pragma unroll
  for (int nt = 0; nt < 8; ++nt) {
    v8f c = {0.f, 0.f, 0.f, 0.f, 0.f, 0.f, 0.f, 0.f};
#pragma unroll
    for (int kt = 0; kt < 4; ++kt) {
      v16h b = *(const v16h*)(w2p + ((long long)(kt * 8 + nt) * 32 + lane) * 16);
      c = __builtin_amdgcn_wmma_f32_16x16x32_f16(false, a[kt].v, false, b,
                                                 (short)0, c, false, false);
    }
    float bias = b2[nt * 16 + M];
    if (active) {
#pragma unroll
      for (int r = 0; r < 8; ++r) {
        int Mr = half ? (r + 8) : r;
        float val = c[r] + bias;
        val = val > 0.f ? val : 0.f;
        int gi = (int)batch[node0 + Mr];
        __hip_atomic_fetch_add(&g[(long long)gi * HIDDEN + nt * 16 + M], val,
                               __ATOMIC_RELAXED, __HIP_MEMORY_SCOPE_AGENT);
      }
    }
  }
}

// ---------------- graph head: relu(g@lin1+b) @ lin2 + b -> log_softmax -----
__global__ void __launch_bounds__(128)
graph_mlp(const float* __restrict__ g,
          const float* __restrict__ l1w, const float* __restrict__ l1b,
          const float* __restrict__ l2w, const float* __restrict__ l2b,
          float* __restrict__ out) {
  __shared__ float sg[HIDDEN];
  __shared__ float sh[HIDDEN];
  __shared__ float sl[N_CLASSES];
  const int gi = blockIdx.x, t = threadIdx.x;
  sg[t] = g[gi * HIDDEN + t];
  __syncthreads();
  float acc = l1b[t];
#pragma unroll 8
  for (int k = 0; k < HIDDEN; ++k) acc += sg[k] * l1w[k * HIDDEN + t];
  sh[t] = acc > 0.f ? acc : 0.f;
  __syncthreads();
  if (t < N_CLASSES) {
    float a2 = l2b[t];
#pragma unroll 8
    for (int k = 0; k < HIDDEN; ++k) a2 += sh[k] * l2w[k * N_CLASSES + t];
    sl[t] = a2;
  }
  __syncthreads();
  if (t < N_CLASSES) {
    float m = sl[0];
#pragma unroll
    for (int j = 1; j < N_CLASSES; ++j) m = fmaxf(m, sl[j]);
    float s = 0.f;
#pragma unroll
    for (int j = 0; j < N_CLASSES; ++j) s += __expf(sl[j] - m);
    out[gi * N_CLASSES + t] = sl[t] - m - __logf(s);
  }
}

// ---------------------------------------------------------------------------
extern "C" void kernel_launch(void* const* d_in, const int* in_sizes, int n_in,
                              void* d_out, int out_size, void* d_ws, size_t ws_size,
                              hipStream_t stream) {
  const float*     x    = (const float*)d_in[0];
  const long long* ei   = (const long long*)d_in[1];
  const long long* bat  = (const long long*)d_in[2];
  const float*     w1   = (const float*)d_in[3];
  const float*     b1   = (const float*)d_in[4];
  const float*     w2   = (const float*)d_in[5];
  const float*     b2   = (const float*)d_in[6];
  const float*     l1w  = (const float*)d_in[7];
  const float*     l1b  = (const float*)d_in[8];
  const float*     l2w  = (const float*)d_in[9];
  const float*     l2b  = (const float*)d_in[10];
  float* out = (float*)d_out;

  char* ws = (char*)d_ws;
  _Float16*  hin = (_Float16*)ws;
  float*     g   = (float*)(ws + OFF_G);
  _Float16*  w1p = (_Float16*)(ws + OFF_W1P);
  _Float16*  w2p = (_Float16*)(ws + OFF_W2P);

  // 1) hin = f16(x) (self term), g = 0
  {
    int nh = N_NODES * F_IN;
    int ng = N_GRAPHS * HIDDEN;
    init_hin_g<<<(nh + 255) / 256, 256, 0, stream>>>(x, hin, g, nh, ng);
  }
  // 2) pack conv weights into WMMA B fragments (f16)
  pack_weights<<<64, 32, 0, stream>>>(w1, w2, w1p, w2p);
  // 3) edge scatter-add via packed-f16 atomics (hin += messages)
  {
    long long nWork = (long long)N_EDGES * 32;
    int blocks = (int)((nWork + 255) / 256);
    edge_scatter_pk<<<blocks, 256, 0, stream>>>((const float4*)x, ei, hin);
  }
  // 4) fused 2-layer node MLP (WMMA) + global_add_pool
  node_mlp<<<(N_TILES + 7) / 8, 256, 0, stream>>>(hin, w1p, w2p, b1, b2, bat, g);
  // 5) graph head + log_softmax
  graph_mlp<<<N_GRAPHS, 128, 0, stream>>>(g, l1w, l1b, l2w, l2b, out);
}
